// TFIng_79431125172782
// MI455X (gfx1250) — matrix operand order repaired and checked by reading, not compiled, not run on hardware
//
#include <hip/hip_runtime.h>
#include <cmath>

// ---------------------------------------------------------------------------
// Types / WMMA helpers (CDNA5 gfx1250, wave32)
// ---------------------------------------------------------------------------
typedef __attribute__((ext_vector_type(16))) _Float16 v16h;
typedef __attribute__((ext_vector_type(8)))  _Float16 v8h;
typedef __attribute__((ext_vector_type(8)))  float    v8f;

#define WMMA_F32F16(A, B, C) \
  __builtin_amdgcn_wmma_f32_16x16x32_f16(false, (A), false, (B), (short)0, (C), false, false)

// Async global->LDS copy (16B per lane), tracked by ASYNCcnt (CDNA5 ISA).
#define ASYNC_LDS_B128(ldsoff, gaddr64) \
  asm volatile("global_load_async_to_lds_b128 %0, %1, off" \
               :: "v"(ldsoff), "v"(gaddr64) : "memory")
#define WAIT_ASYNC0() asm volatile("s_wait_asynccnt 0" ::: "memory")

// A-fragment (16x32 f16): lane holds row m = lane&15; element e at
// k = e + 16*half (two contiguous 8-half runs).
static __device__ __forceinline__ v16h ld_a_frag(const _Float16* p0, const _Float16* p1) {
  union { v16h v; v8h h[2]; } u;
  u.h[0] = *(const v8h*)p0;
  u.h[1] = *(const v8h*)(p1);
  return u.v;
}
// B-fragment (32x16 f16): lane holds col n = lane&15; element e at k = e + 16*half.
static __device__ __forceinline__ v16h ld_b_frag(const _Float16* p) {
  union { v16h v; v8h h[2]; } u;
  u.h[0] = *(const v8h*)p;
  u.h[1] = *(const v8h*)(p + 8);
  return u.v;
}
static __device__ __forceinline__ v8f vzero8() {
  v8f z;
  #pragma unroll
  for (int i = 0; i < 8; ++i) z[i] = 0.0f;
  return z;
}

// ---------------------------------------------------------------------------
// Tiled WMMA GEMM, f16 operands, async double-buffered LDS staging.
//   Cf/Ch[M,N] = A[M,K](f16) @ Bt[N,K]^T(f16)  (+bias) (relu) (+resid f32)
// Block: 256 threads = 8 waves; tile 128x128, BK=32.
// ---------------------------------------------------------------------------
template<int BIAS, int RELU, int RESID, int OUTH>
__global__ __launch_bounds__(256)
void gemm_wmma(const _Float16* __restrict__ A, const _Float16* __restrict__ Bt,
               const float* __restrict__ bias, const float* __restrict__ resid,
               float* __restrict__ Cf, _Float16* __restrict__ Ch,
               int M, int N, int K)
{
  // pitch 40 halves = 80B: keeps every 16B chunk and every fragment 16B-aligned.
  __shared__ __align__(16) _Float16 lA[2][128][40];
  __shared__ __align__(16) _Float16 lB[2][128][40];   // lB[buf][n][k]

  const int t    = threadIdx.x;
  const int wave = t >> 5;
  const int lane = t & 31;
  const int hgrp = lane >> 4;
  const int lr   = lane & 15;
  const int wm   = wave & 3;
  const int wn   = wave >> 2;
  const int m0   = blockIdx.y * 128;
  const int n0   = blockIdx.x * 128;

  const unsigned baseA = (unsigned)(uintptr_t)&lA[0][0][0];
  const unsigned baseB = (unsigned)(uintptr_t)&lB[0][0][0];
  const unsigned BUFSZ = 128u * 40u * 2u;             // 10240 bytes

  // Issue one 128x32 A-slab + 32x128 B-slab (as Bt rows) into LDS buffer `buf`.
  auto issue_slab = [&](int k0, int buf) {
    #pragma unroll
    for (int i = 0; i < 2; ++i) {
      int cid = t + 256 * i;            // 512 chunks of 16B per tile
      int r = cid >> 2, c = cid & 3;
      unsigned lo = baseA + (unsigned)buf * BUFSZ + (unsigned)r * 80u + (unsigned)c * 16u;
      unsigned long long ga =
          (unsigned long long)(const void*)(A + (size_t)(m0 + r) * K + k0 + c * 8);
      ASYNC_LDS_B128(lo, ga);
    }
    #pragma unroll
    for (int i = 0; i < 2; ++i) {
      int cid = t + 256 * i;
      int r = cid >> 2, c = cid & 3;
      unsigned lo = baseB + (unsigned)buf * BUFSZ + (unsigned)r * 80u + (unsigned)c * 16u;
      unsigned long long ga =
          (unsigned long long)(const void*)(Bt + (size_t)(n0 + r) * K + k0 + c * 8);
      ASYNC_LDS_B128(lo, ga);
    }
  };

  v8f acc[2][4];
  #pragma unroll
  for (int i = 0; i < 2; ++i)
    #pragma unroll
    for (int j = 0; j < 4; ++j) acc[i][j] = vzero8();

  const int nslab = K >> 5;
  issue_slab(0, 0);
  for (int s = 0; s < nslab; ++s) {
    WAIT_ASYNC0();        // this wave's slab-s copies have landed in LDS
    __syncthreads();      // ... and so have every other wave's
    if (s + 1 < nslab) issue_slab((s + 1) << 5, (s + 1) & 1);  // stream next slab
    const int buf = s & 1;

    v16h af[2], bf[4];
    #pragma unroll
    for (int i = 0; i < 2; ++i) {
      const _Float16* rp = &lA[buf][wm * 32 + i * 16 + lr][0];
      af[i] = ld_a_frag(rp + 16 * hgrp, rp + 8 + 16 * hgrp);
    }
    #pragma unroll
    for (int j = 0; j < 4; ++j) {
      const _Float16* cp = &lB[buf][wn * 64 + j * 16 + lr][0];
      bf[j] = ld_b_frag(cp + 16 * hgrp);
    }
    #pragma unroll
    for (int i = 0; i < 2; ++i)
      #pragma unroll
      for (int j = 0; j < 4; ++j)
        acc[i][j] = WMMA_F32F16(af[i], bf[j], acc[i][j]);
    __syncthreads();      // all waves done reading buf before it is refilled
  }

  // Epilogue: C layout -> VGPR r holds row m = r + 8*hgrp, lane lr holds col.
  #pragma unroll
  for (int i = 0; i < 2; ++i) {
    #pragma unroll
    for (int j = 0; j < 4; ++j) {
      int gn = n0 + wn * 64 + j * 16 + lr;
      float bv = BIAS ? bias[gn] : 0.0f;
      #pragma unroll
      for (int r = 0; r < 8; ++r) {
        int gm = m0 + wm * 32 + i * 16 + r + 8 * hgrp;
        float v = acc[i][j][r] + bv;
        if (RELU) v = fmaxf(v, 0.0f);
        if (RESID) v += resid[(size_t)gm * 512 + gn];   // residual width always D=512
        if (OUTH) Ch[(size_t)gm * N + gn] = (_Float16)v;
        else      Cf[(size_t)gm * N + gn] = v;
      }
    }
  }
}

// ---------------------------------------------------------------------------
// Fused attention, one (b,h) per block; f16 in, f16 out. S=64, dh=64, H=8.
// ---------------------------------------------------------------------------
__global__ __launch_bounds__(128)
void attn_wmma(const _Float16* __restrict__ Q, const _Float16* __restrict__ K,
               const _Float16* __restrict__ V, _Float16* __restrict__ O, int causal)
{
  __shared__ __align__(16) _Float16 lQ [64][72];
  __shared__ __align__(16) _Float16 lK [64][72];
  __shared__ __align__(16) _Float16 lVt[64][72];  // V transposed: lVt[d][s]
  __shared__ __align__(16) _Float16 lP [64][72];

  const int t    = threadIdx.x;
  const int wave = t >> 5;
  const int lane = t & 31;
  const int hgrp = lane >> 4;
  const int lr   = lane & 15;
  const int b    = blockIdx.x >> 3;
  const int h    = blockIdx.x & 7;
  const size_t base = ((size_t)b * 64) * 512 + (size_t)h * 64;

  #pragma unroll
  for (int i = 0; i < 32; ++i) {
    int id = t + 128 * i;
    int s = id >> 6, d = id & 63;
    lQ[s][d]  = Q[base + (size_t)s * 512 + d];
    lK[s][d]  = K[base + (size_t)s * 512 + d];
    lVt[d][s] = V[base + (size_t)s * 512 + d];
  }
  __syncthreads();

  // S = Q @ K^T (K rows are K-major for the B-fragment -> contiguous reads)
  v8f sacc[4];
  #pragma unroll
  for (int j = 0; j < 4; ++j) sacc[j] = vzero8();
  #pragma unroll
  for (int k0 = 0; k0 < 64; k0 += 32) {
    const _Float16* rp = &lQ[wave * 16 + lr][0];
    v16h aq = ld_a_frag(rp + k0 + 16 * hgrp, rp + k0 + 8 + 16 * hgrp);
    #pragma unroll
    for (int j = 0; j < 4; ++j) {
      v16h bk = ld_b_frag(&lK[j * 16 + lr][k0 + 16 * hgrp]);
      sacc[j] = WMMA_F32F16(aq, bk, sacc[j]);
    }
  }

  // scale + causal mask + exact softmax per row (16-lane shfl reductions).
  const float scale = 0.125f;  // 1/sqrt(64)
  #pragma unroll
  for (int r = 0; r < 8; ++r) {
    int m = wave * 16 + r + 8 * hgrp;
    float vals[4];
    float mx = -3.0e38f;
    #pragma unroll
    for (int j = 0; j < 4; ++j) {
      int n = j * 16 + lr;
      float s = sacc[j][r] * scale;
      if (causal && (n > m)) s = -1.0e9f;
      vals[j] = s;
      mx = fmaxf(mx, s);
    }
    #pragma unroll
    for (int off = 1; off < 16; off <<= 1) mx = fmaxf(mx, __shfl_xor(mx, off));
    float sum = 0.0f;
    #pragma unroll
    for (int j = 0; j < 4; ++j) { vals[j] = expf(vals[j] - mx); sum += vals[j]; }
    #pragma unroll
    for (int off = 1; off < 16; off <<= 1) sum += __shfl_xor(sum, off);
    float inv = 1.0f / sum;
    #pragma unroll
    for (int j = 0; j < 4; ++j) lP[m][j * 16 + lr] = (_Float16)(vals[j] * inv);
  }
  __syncthreads();

  // O = P @ V (Vt rows are K-major for the B-fragment)
  v8f oacc[4];
  #pragma unroll
  for (int j = 0; j < 4; ++j) oacc[j] = vzero8();
  #pragma unroll
  for (int k0 = 0; k0 < 64; k0 += 32) {
    const _Float16* rp = &lP[wave * 16 + lr][0];
    v16h ap = ld_a_frag(rp + k0 + 16 * hgrp, rp + k0 + 8 + 16 * hgrp);
    #pragma unroll
    for (int j = 0; j < 4; ++j) {
      v16h bv = ld_b_frag(&lVt[j * 16 + lr][k0 + 16 * hgrp]);
      oacc[j] = WMMA_F32F16(ap, bv, oacc[j]);
    }
  }
  #pragma unroll
  for (int j = 0; j < 4; ++j)
    #pragma unroll
    for (int r = 0; r < 8; ++r)
      O[base + (size_t)(wave * 16 + r + 8 * hgrp) * 512 + (j * 16 + lr)] =
          (_Float16)oacc[j][r];
}

// ---------------------------------------------------------------------------
// LayerNorm over D=512, one row per wave; dual f32 (residual) + f16 (GEMM A) out.
// ---------------------------------------------------------------------------
__global__ __launch_bounds__(256)
void layernorm_k(const float* __restrict__ X, const float* __restrict__ g,
                 const float* __restrict__ be, float* __restrict__ Yf,
                 _Float16* __restrict__ Yh)
{
  const int wave = threadIdx.x >> 5, lane = threadIdx.x & 31;
  const int row  = blockIdx.x * 8 + wave;
  const float* x = X + (size_t)row * 512;
  float v[16], s = 0.0f, sq = 0.0f;
  #pragma unroll
  for (int e = 0; e < 16; ++e) {
    float xv = x[lane + 32 * e];
    v[e] = xv; s += xv; sq += xv * xv;
  }
  #pragma unroll
  for (int off = 1; off < 32; off <<= 1) { s += __shfl_xor(s, off); sq += __shfl_xor(sq, off); }
  float mu  = s * (1.0f / 512.0f);
  float var = sq * (1.0f / 512.0f) - mu * mu;
  float rs  = rsqrtf(var + 1e-3f);
  float*     yf = Yf + (size_t)row * 512;
  _Float16*  yh = Yh + (size_t)row * 512;
  #pragma unroll
  for (int e = 0; e < 16; ++e) {
    int c = lane + 32 * e;
    float o = (v[e] - mu) * rs * g[c] + be[c];
    yf[c] = o;
    yh[c] = (_Float16)o;
  }
}

// ---------------------------------------------------------------------------
// x = tok_emb[targets] + pos_emb  (dual f32 + f16)
// ---------------------------------------------------------------------------
__global__ __launch_bounds__(256)
void embed_k(const int* __restrict__ targets, const float* __restrict__ tok,
             const float* __restrict__ pos, float* __restrict__ Xf,
             _Float16* __restrict__ Xh)
{
  int gid = blockIdx.x * 256 + threadIdx.x;   // 4096*512 total
  int row = gid >> 9, d = gid & 511;
  int s = row & 63;
  int tk = targets[row];
  float v = tok[(size_t)tk * 512 + d] + pos[(size_t)s * 512 + d];
  Xf[gid] = v;
  Xh[gid] = (_Float16)v;
}

// f32 -> f16 elementwise (enc_feat staging)
__global__ __launch_bounds__(256)
void cvt_k(const float* __restrict__ src, _Float16* __restrict__ dst)
{
  int gid = blockIdx.x * 256 + threadIdx.x;
  dst[gid] = (_Float16)src[gid];
}

// Transpose-convert: W[K,N] f32 -> Wt[N,K] f16 (32x32 LDS tiles).
__global__ __launch_bounds__(256)
void cvtT_k(const float* __restrict__ W, _Float16* __restrict__ Wt, int K, int N)
{
  __shared__ float tile[32][33];
  const int n0 = blockIdx.x * 32, k0 = blockIdx.y * 32;
  const int x = threadIdx.x & 31, y = (threadIdx.x >> 5) * 4;  // 256 thr = 32x8x(4 rows)
  #pragma unroll
  for (int j = 0; j < 4; ++j)
    tile[y + j][x] = W[(size_t)(k0 + y + j) * N + (n0 + x)];
  __syncthreads();
  #pragma unroll
  for (int j = 0; j < 4; ++j)
    Wt[(size_t)(n0 + y + j) * K + (k0 + x)] = (_Float16)tile[x][y + j];
}

// ---------------------------------------------------------------------------
// Replacement mask: out[b,i,v] = -inf iff some j<=i has targets[b,j]==v, v!=0.
// ---------------------------------------------------------------------------
__global__ __launch_bounds__(256)
void mask_k(const int* __restrict__ targets, float* __restrict__ out)
{
  __shared__ unsigned seen[256];
  const int b = blockIdx.x, t = threadIdx.x;
  seen[t] = 0u;
  __syncthreads();
  const float ninf = -__builtin_inff();
  for (int i = 0; i < 64; ++i) {
    if (t == 0) {
      int tk = targets[b * 64 + i];
      if (tk != 0) seen[tk >> 5] |= (1u << (tk & 31));
    }
    __syncthreads();
    float* row = out + ((size_t)b * 64 + i) * 8192;
    #pragma unroll
    for (int c = 0; c < 32; ++c) {
      int v = t + 256 * c;
      if (seen[v >> 5] & (1u << (v & 31))) row[v] = ninf;
    }
    __syncthreads();
  }
}

// ---------------------------------------------------------------------------
// Host orchestration
// ---------------------------------------------------------------------------
extern "C" void kernel_launch(void* const* d_in, const int* in_sizes, int n_in,
                              void* d_out, int out_size, void* d_ws, size_t ws_size,
                              hipStream_t stream)
{
  (void)in_sizes; (void)n_in; (void)out_size; (void)ws_size;

  const float* enc_feat = (const float*)d_in[0];
  const int*   targets  = (const int*)  d_in[1];
  const float* conv_w   = (const float*)d_in[2];
  const float* conv_b   = (const float*)d_in[3];
  const float* tok_emb  = (const float*)d_in[4];
  const float* pos_emb  = (const float*)d_in[5];
  const float* Wq_s = (const float*)d_in[6];
  const float* Wk_s = (const float*)d_in[7];
  const float* Wv_s = (const float*)d_in[8];
  const float* Wo_s = (const float*)d_in[9];
  const float* Wq_c = (const float*)d_in[10];
  const float* Wk_c = (const float*)d_in[11];
  const float* Wv_c = (const float*)d_in[12];
  const float* Wo_c = (const float*)d_in[13];
  const float* W1   = (const float*)d_in[14];
  const float* fb1  = (const float*)d_in[15];
  const float* W2   = (const float*)d_in[16];
  const float* fb2  = (const float*)d_in[17];
  const float* g1   = (const float*)d_in[18];
  const float* be1  = (const float*)d_in[19];
  const float* g2   = (const float*)d_in[20];
  const float* be2  = (const float*)d_in[21];
  const float* g3   = (const float*)d_in[22];
  const float* be3  = (const float*)d_in[23];
  const float* Wout = (const float*)d_in[24];
  const float* bout = (const float*)d_in[25];
  float* out = (float*)d_out;

  const int M = 64 * 64;            // 4096 rows (B*S == B*64 enc pixels)
  const int Dm = 512, Ff = 2048, Vv = 8192, Cc = 2048, Ll = 6;
  const size_t SZ = (size_t)M * Dm;
  const size_t DD = (size_t)Dm * Dm, DF = (size_t)Dm * Ff;

  // ---- workspace carve-up ----
  char* p = (char*)d_ws;
  float*    X    = (float*)p;     p += SZ * 4;
  float*    Tb   = (float*)p;     p += SZ * 4;
  _Float16* Xh   = (_Float16*)p;  p += SZ * 2;
  _Float16* ENCh = (_Float16*)p;  p += SZ * 2;
  _Float16* Qh   = (_Float16*)p;  p += SZ * 2;
  _Float16* Kh   = (_Float16*)p;  p += SZ * 2;
  _Float16* Vh   = (_Float16*)p;  p += SZ * 2;
  _Float16* AOh  = (_Float16*)p;  p += SZ * 2;
  _Float16* HFh  = (_Float16*)p;  p += (size_t)M * Ff * 2;
  _Float16* EFh  = (_Float16*)p;  p += (size_t)M * Cc * 2;
  _Float16* convT = (_Float16*)p; p += (size_t)Cc * Dm * 2;
  _Float16* WoutT = (_Float16*)p; p += (size_t)Dm * Vv * 2;
  _Float16* layT  = (_Float16*)p; // per-layer: 8*DD + 2*DF halves
  const size_t LST = 8 * DD + 2 * DF;

  const dim3 blk(256);
  const dim3 gD(Dm / 128, M / 128);
  const dim3 gF(Ff / 128, M / 128);
  const dim3 gV(Vv / 128, M / 128);

  // ---- one-time (per call) f16 staging: weights transposed, enc_feat ----
  cvt_k<<<(M * Cc) / 256, blk, 0, stream>>>(enc_feat, EFh);
  cvtT_k<<<dim3(Dm / 32, Cc / 32), blk, 0, stream>>>(conv_w, convT, Cc, Dm);
  cvtT_k<<<dim3(Vv / 32, Dm / 32), blk, 0, stream>>>(Wout, WoutT, Dm, Vv);
  for (int l = 0; l < Ll; ++l) {
    _Float16* wt = layT + (size_t)l * LST;
    const dim3 gw(Dm / 32, Dm / 32);
    cvtT_k<<<gw, blk, 0, stream>>>(Wq_s + l * DD, wt + 0 * DD, Dm, Dm);
    cvtT_k<<<gw, blk, 0, stream>>>(Wk_s + l * DD, wt + 1 * DD, Dm, Dm);
    cvtT_k<<<gw, blk, 0, stream>>>(Wv_s + l * DD, wt + 2 * DD, Dm, Dm);
    cvtT_k<<<gw, blk, 0, stream>>>(Wo_s + l * DD, wt + 3 * DD, Dm, Dm);
    cvtT_k<<<gw, blk, 0, stream>>>(Wq_c + l * DD, wt + 4 * DD, Dm, Dm);
    cvtT_k<<<gw, blk, 0, stream>>>(Wk_c + l * DD, wt + 5 * DD, Dm, Dm);
    cvtT_k<<<gw, blk, 0, stream>>>(Wv_c + l * DD, wt + 6 * DD, Dm, Dm);
    cvtT_k<<<gw, blk, 0, stream>>>(Wo_c + l * DD, wt + 7 * DD, Dm, Dm);
    cvtT_k<<<dim3(Ff / 32, Dm / 32), blk, 0, stream>>>(W1 + (size_t)l * DF, wt + 8 * DD, Dm, Ff);
    cvtT_k<<<dim3(Dm / 32, Ff / 32), blk, 0, stream>>>(W2 + (size_t)l * DF, wt + 8 * DD + DF, Ff, Dm);
  }

  // ---- forward pass ----
  embed_k<<<(M * Dm) / 256, blk, 0, stream>>>(targets, tok_emb, pos_emb, X, Xh);
  gemm_wmma<1,0,0,1><<<gD, blk, 0, stream>>>(EFh, convT, conv_b, nullptr,
                                             nullptr, ENCh, M, Dm, Cc);

  for (int l = 0; l < Ll; ++l) {
    _Float16* wt = layT + (size_t)l * LST;
    // ---- causal self-attention ----
    gemm_wmma<0,0,0,1><<<gD, blk, 0, stream>>>(Xh, wt + 0 * DD, nullptr, nullptr,
                                               nullptr, Qh, M, Dm, Dm);
    gemm_wmma<0,0,0,1><<<gD, blk, 0, stream>>>(Xh, wt + 1 * DD, nullptr, nullptr,
                                               nullptr, Kh, M, Dm, Dm);
    gemm_wmma<0,0,0,1><<<gD, blk, 0, stream>>>(Xh, wt + 2 * DD, nullptr, nullptr,
                                               nullptr, Vh, M, Dm, Dm);
    attn_wmma<<<64 * 8, dim3(128), 0, stream>>>(Qh, Kh, Vh, AOh, 1);
    gemm_wmma<0,0,1,0><<<gD, blk, 0, stream>>>(AOh, wt + 3 * DD, nullptr, X,
                                               Tb, nullptr, M, Dm, Dm);
    layernorm_k<<<M / 8, blk, 0, stream>>>(Tb, g1 + l * Dm, be1 + l * Dm, X, Xh);
    // ---- cross-attention vs encoder ----
    gemm_wmma<0,0,0,1><<<gD, blk, 0, stream>>>(Xh,   wt + 4 * DD, nullptr, nullptr,
                                               nullptr, Qh, M, Dm, Dm);
    gemm_wmma<0,0,0,1><<<gD, blk, 0, stream>>>(ENCh, wt + 5 * DD, nullptr, nullptr,
                                               nullptr, Kh, M, Dm, Dm);
    gemm_wmma<0,0,0,1><<<gD, blk, 0, stream>>>(ENCh, wt + 6 * DD, nullptr, nullptr,
                                               nullptr, Vh, M, Dm, Dm);
    attn_wmma<<<64 * 8, dim3(128), 0, stream>>>(Qh, Kh, Vh, AOh, 0);
    gemm_wmma<0,0,1,0><<<gD, blk, 0, stream>>>(AOh, wt + 7 * DD, nullptr, X,
                                               Tb, nullptr, M, Dm, Dm);
    layernorm_k<<<M / 8, blk, 0, stream>>>(Tb, g2 + l * Dm, be2 + l * Dm, X, Xh);
    // ---- FFN ----
    gemm_wmma<1,1,0,1><<<gF, blk, 0, stream>>>(Xh, wt + 8 * DD, fb1 + (size_t)l * Ff,
                                               nullptr, nullptr, HFh, M, Ff, Dm);
    gemm_wmma<1,0,1,0><<<gD, blk, 0, stream>>>(HFh, wt + 8 * DD + DF, fb2 + (size_t)l * Dm,
                                               X, Tb, nullptr, M, Dm, Ff);
    layernorm_k<<<M / 8, blk, 0, stream>>>(Tb, g3 + l * Dm, be3 + l * Dm, X, Xh);
  }

  // logits = x @ Wout + bout, then replacement mask.
  gemm_wmma<1,0,0,0><<<gV, blk, 0, stream>>>(Xh, WoutT, bout, nullptr,
                                             out, nullptr, M, Vv, Dm);
  mask_k<<<64, blk, 0, stream>>>(targets, out);
}
